// LiltSelfAttention_55336358642313
// MI455X (gfx1250) — matrix operand order repaired and checked by reading, not compiled, not run on hardware
//
#include <hip/hip_runtime.h>

// ---------------------------------------------------------------------------
// LiLT self-attention, fused for MI455X (gfx1250, wave32, WMMA bf16 + TDM)
// ---------------------------------------------------------------------------
#define H_   12
#define S_   1024
#define B_   8
#define D_   768
#define DL_  192
#define AUG  96     // augmented QK head dim: 64 text + 16 layout + 16 zero pad
#define VD   80     // augmented V head dim: 64 text + 16 layout
#define NKB  (S_ / 64)

typedef __bf16 bhalf;
typedef __attribute__((ext_vector_type(16))) bhalf v16bf;
typedef __attribute__((ext_vector_type(8)))  float v8f;
typedef __attribute__((ext_vector_type(4)))  unsigned int v4u_;
typedef __attribute__((ext_vector_type(8)))  int v8i_;
typedef __attribute__((ext_vector_type(4)))  int v4i_;

union FragU { v16bf v; uint4 q[2]; };

// Tensor Data Mover availability (device pass only; host pass sees 0)
#if defined(__has_builtin)
#  if __has_builtin(__builtin_amdgcn_tensor_load_to_lds) && \
      __has_builtin(__builtin_amdgcn_s_wait_tensorcnt)
#    define USE_TDM 1
#  endif
#endif
#ifndef USE_TDM
#  define USE_TDM 0
#endif

__device__ __forceinline__ unsigned short f2bf(float f) {
    unsigned int u = __float_as_uint(f);
    unsigned int r = u + 0x7FFFu + ((u >> 16) & 1u);   // round-to-nearest-even
    return (unsigned short)(r >> 16);
}

__device__ __forceinline__ v8f wmma_bf16(const FragU& a, const FragU& b, v8f c) {
    return __builtin_amdgcn_wmma_f32_16x16x32_bf16(
        false, a.v, false, b.v, (short)0, c, false, false);
}

#if USE_TDM
// Issue one 2D TDM tile load (bf16 elements) into LDS. All dims/strides in
// elements (data_size = 2 bytes). D# layout per CDNA5 ISA ch.8.
__device__ __forceinline__ void tdm_load_2d(unsigned lds_addr, const void* gptr,
                                            unsigned tile_d0, unsigned tile_d1,
                                            unsigned tensor_d0, unsigned tensor_d1,
                                            unsigned stride_d0) {
    const size_t ga = (size_t)gptr;
    v4u_ g0;
    g0[0] = 1u;                                        // count=1, user mode
    g0[1] = lds_addr;                                  // lds_addr       [63:32]
    g0[2] = (unsigned)(ga & 0xFFFFFFFFu);              // global_addr lo [95:64]
    g0[3] = (unsigned)((ga >> 32) & 0x01FFFFFFu)       // global_addr hi [120:96]
            | (2u << 30);                              // type=2 (image) [127:126]
    v8i_ g1;
    g1[0] = (int)(1u << 16);                           // data_size=1 (2 bytes)
    g1[1] = (int)((tensor_d0 & 0xFFFFu) << 16);        // tensor_dim0 [79:48]
    g1[2] = (int)(((tensor_d0 >> 16) & 0xFFFFu) |
                  ((tensor_d1 & 0xFFFFu) << 16));      // + tensor_dim1 [111:80]
    g1[3] = (int)(((tensor_d1 >> 16) & 0xFFFFu) |
                  (tile_d0 << 16));                    // tile_dim0 [127:112]
    g1[4] = (int)tile_d1;                              // tile_dim1, tile_dim2=0
    g1[5] = (int)stride_d0;                            // tensor_dim0_stride lo
    g1[6] = 0;                                         // stride hi, dim1_stride
    g1[7] = 0;
    v4i_ gz = (v4i_){0, 0, 0, 0};
#if __has_include(<hip/amd_detail/amd_gfx1250_TDM.h>)
    v8i_ gz8 = (v8i_){0, 0, 0, 0, 0, 0, 0, 0};
    __builtin_amdgcn_tensor_load_to_lds(g0, g1, gz, gz, gz8, 0);
#else
    __builtin_amdgcn_tensor_load_to_lds(g0, g1, gz, gz, 0);
#endif
}
#endif

// ---------------------------------------------------------------------------
// f32 -> bf16 conversion / zero fill
// ---------------------------------------------------------------------------
__global__ void cvt_bf16_kernel(const float* __restrict__ in,
                                unsigned short* __restrict__ out, int n) {
    int i = blockIdx.x * blockDim.x + threadIdx.x;
    if (i < n) out[i] = f2bf(in[i]);
}

__global__ void zero_u32_kernel(unsigned int* __restrict__ p, int n) {
    int i = blockIdx.x * blockDim.x + threadIdx.x;
    if (i < n) p[i] = 0u;
}

// ---------------------------------------------------------------------------
// Projection GEMM:  val = (X[M,K] * W[N,K]^T + bias) * scale, scattered into
//   out[(b*H+h)*head_block + (out_off+d)*dim_stride + s*row_stride]
// QA/KA: row-major augmented rows (dim_stride=1,row_stride=AUG)
// V:     dim-major per head     (dim_stride=S, row_stride=1)   -> TDM friendly
// ---------------------------------------------------------------------------
__global__ void gemm_proj_kernel(const unsigned short* __restrict__ X, int K,
                                 const unsigned short* __restrict__ W,
                                 const float* __restrict__ bias,
                                 unsigned short* __restrict__ out,
                                 int head_dim, int dim_stride, int row_stride,
                                 int head_block, int out_off, float scale) {
    const int lane  = threadIdx.x & 31;
    const int wave  = threadIdx.x >> 5;
    const int Rbase = blockIdx.x * 128 + wave * 16;
    const int N0    = blockIdx.y * 64;
    const int klo8  = (lane >> 4) * 8;    // A-fragment K swizzle
    const int khalf = (lane >> 4) * 16;   // unused; B fragments use klo8 too
    const int col16 = lane & 15;

    const unsigned short* Xrow = X + (size_t)(Rbase + col16) * K;

    v8f acc[4];
#pragma unroll
    for (int t = 0; t < 4; ++t) acc[t] = (v8f){};

    for (int k0 = 0; k0 < K; k0 += 32) {
        FragU a;
        a.q[0] = *(const uint4*)&Xrow[k0 + klo8];
        a.q[1] = *(const uint4*)&Xrow[k0 + 16 + klo8];
#pragma unroll
        for (int t = 0; t < 4; ++t) {
            FragU bw;
            const unsigned short* Wr =
                W + (size_t)(N0 + t * 16 + col16) * K + k0 + klo8;
            bw.q[0] = *(const uint4*)&Wr[0];
            bw.q[1] = *(const uint4*)&Wr[16];
            acc[t] = wmma_bf16(a, bw, acc[t]);
        }
    }

#pragma unroll
    for (int t = 0; t < 4; ++t) {
        const int n  = N0 + t * 16 + col16;
        const float bv = bias[n];
        const int hh = n / head_dim;
        const int d  = n % head_dim;
#pragma unroll
        for (int r = 0; r < 8; ++r) {
            const int row = Rbase + r + ((lane >> 4) << 3);
            const int bb  = row >> 10;         // row = b*1024 + s
            const int ss  = row & 1023;
            const float v = (acc[t][r] + bv) * scale;
            out[(size_t)(bb * H_ + hh) * head_block +
                (size_t)(out_off + d) * dim_stride + (size_t)ss * row_stride] =
                f2bf(v);
        }
    }
}

// ---------------------------------------------------------------------------
// Fused flash attention over augmented heads.
// grid = (B*H, S/128); block = 256 (8 waves x 16 query rows).
// 64-key blocks double-buffered in LDS via TDM (fallback: sync b128 copies).
// ---------------------------------------------------------------------------
__global__ void attn_kernel(const unsigned short* __restrict__ QA,
                            const unsigned short* __restrict__ KA,
                            const unsigned short* __restrict__ VA,
                            float* __restrict__ ctx,
                            float* __restrict__ lctx) {
    __shared__ __align__(16) unsigned short sK[2][64 * AUG];   // 2 x 12 KB
    __shared__ __align__(16) unsigned short sVT[2][VD * 64];   // 2 x 10 KB
    __shared__ __align__(16) unsigned short sP[8 * 16 * 64];   // 16 KB

    const int tid   = threadIdx.x;
    const int lane  = tid & 31;
    const int wave  = tid >> 5;
    const int bh    = blockIdx.x;
    const int b     = bh / H_;
    const int h     = bh % H_;
    const int col16 = lane & 15;
    const int klo8  = (lane >> 4) * 8;
    const int khalf = (lane >> 4) * 16;

    const unsigned short* Qp = QA + (size_t)bh * S_ * AUG;
    const unsigned short* Kp = KA + (size_t)bh * S_ * AUG;
    const unsigned short* Vp = VA + (size_t)bh * VD * S_;   // dim-major [80][1024]

    // resident Q fragments (this wave's 16 query rows, K = 96 = 3 x 32)
    const int qs = blockIdx.y * 128 + wave * 16 + col16;
    const unsigned short* Qrow = Qp + (size_t)qs * AUG;
    FragU aq[3];
#pragma unroll
    for (int c = 0; c < 3; ++c) {
        aq[c].q[0] = *(const uint4*)&Qrow[c * 32 + klo8];
        aq[c].q[1] = *(const uint4*)&Qrow[c * 32 + 16 + klo8];
    }

    v8f o[5];
#pragma unroll
    for (int t = 0; t < 5; ++t) o[t] = (v8f){};
    float m8[8], l8[8];
#pragma unroll
    for (int r = 0; r < 8; ++r) { m8[r] = -3.0e38f; l8[r] = 0.f; }

    unsigned short* Pw = &sP[wave * 16 * 64];

#if USE_TDM
    // prologue: DMA block 0 into buffer 0 (wave 0 issues; EXEC-independent op)
    if (wave == 0) {
        tdm_load_2d((unsigned)(size_t)&sK[0][0],  Kp, AUG, 64, AUG, 64, AUG);
        tdm_load_2d((unsigned)(size_t)&sVT[0][0], Vp, 64, VD, 64, VD, S_);
    }
#endif

    for (int kb = 0; kb < NKB; ++kb) {
        const int cur = kb & 1;
#if USE_TDM
        if (wave == 0) {
            if (kb + 1 < NKB) {   // overlap next block's DMA with this compute
                const int nxt = cur ^ 1;
                tdm_load_2d((unsigned)(size_t)&sK[nxt][0],
                            Kp + (size_t)(kb + 1) * 64 * AUG,
                            AUG, 64, AUG, 64, AUG);
                tdm_load_2d((unsigned)(size_t)&sVT[nxt][0],
                            Vp + (size_t)(kb + 1) * 64,
                            64, VD, 64, VD, S_);
                __builtin_amdgcn_s_wait_tensorcnt(2);  // current pair done
            } else {
                __builtin_amdgcn_s_wait_tensorcnt(0);
            }
        }
        __syncthreads();
#else
        // synchronous staging fallback
        for (int i = tid; i < 64 * AUG / 8; i += 256) {
            int key = i / 12, ch = i % 12;
            *(uint4*)&sK[cur][key * AUG + ch * 8] =
                *(const uint4*)&Kp[(size_t)(kb * 64 + key) * AUG + ch * 8];
        }
        for (int i = tid; i < VD * 64 / 8; i += 256) {
            int dim = i >> 3, ch = i & 7;
            *(uint4*)&sVT[cur][dim * 64 + ch * 8] =
                *(const uint4*)&Vp[(size_t)dim * S_ + kb * 64 + ch * 8];
        }
        __syncthreads();
#endif

        // ---- S = (Q_aug * K_aug^T) / 8 ----
        v8f s4[4];
#pragma unroll
        for (int ct = 0; ct < 4; ++ct) {
            v8f acc = (v8f){};
            const int keycol = ct * 16 + col16;
#pragma unroll
            for (int c = 0; c < 3; ++c) {
                FragU bk;
                bk.q[0] = *(uint4*)&sK[cur][keycol * AUG + c * 32 + klo8];
                bk.q[1] = *(uint4*)&sK[cur][keycol * AUG + c * 32 + 16 + klo8];
                acc = wmma_bf16(aq[c], bk, acc);
            }
            s4[ct] = acc;
        }

        // ---- online softmax: row reductions across 16-lane halves ----
#pragma unroll
        for (int r = 0; r < 8; ++r) {
            float mx = -3.0e38f;
#pragma unroll
            for (int ct = 0; ct < 4; ++ct) {
                s4[ct][r] *= 0.125f;
                mx = fmaxf(mx, s4[ct][r]);
            }
#pragma unroll
            for (int d = 1; d < 16; d <<= 1)
                mx = fmaxf(mx, __shfl_xor(mx, d, 32));
            const float mnew = fmaxf(m8[r], mx);
            const float c0 = __expf(m8[r] - mnew);
            m8[r] = mnew;
            float rs = 0.f;
#pragma unroll
            for (int ct = 0; ct < 4; ++ct) {
                const float p = __expf(s4[ct][r] - mnew);
                s4[ct][r] = p;
                rs += p;
            }
#pragma unroll
            for (int d = 1; d < 16; d <<= 1) rs += __shfl_xor(rs, d, 32);
            l8[r] = l8[r] * c0 + rs;
#pragma unroll
            for (int t = 0; t < 5; ++t) o[t][r] *= c0;
        }

        // ---- P -> LDS (bf16): re-swizzle C layout into A-fragment layout ----
        const int mh = (lane >> 4) * 8;
#pragma unroll
        for (int ct = 0; ct < 4; ++ct)
#pragma unroll
            for (int r = 0; r < 8; ++r)
                Pw[(r + mh) * 64 + ct * 16 + col16] = f2bf(s4[ct][r]);
        // same-wave LDS RAW: DS ops in-order within a wave

        FragU pa[2];
#pragma unroll
        for (int f = 0; f < 2; ++f) {
            pa[f].q[0] = *(uint4*)&Pw[col16 * 64 + f * 32 + klo8];
            pa[f].q[1] = *(uint4*)&Pw[col16 * 64 + f * 32 + 16 + klo8];
        }

        // ---- O += P * V_aug (V staged dim-major => contiguous B frags) ----
#pragma unroll
        for (int t = 0; t < 5; ++t) {
            v8f acc = o[t];
            const int dimc = t * 16 + col16;
#pragma unroll
            for (int f = 0; f < 2; ++f) {
                FragU bv;
                bv.q[0] = *(uint4*)&sVT[cur][dimc * 64 + f * 32 + klo8];
                bv.q[1] = *(uint4*)&sVT[cur][dimc * 64 + f * 32 + 16 + klo8];
                acc = wmma_bf16(pa[f], bv, acc);
            }
            o[t] = acc;
        }
        __syncthreads();   // all reads of buf[cur] done; next DMA may reuse it
    }

    // ---- normalize and scatter (ctx f32 [B,S,H*64], lctx f32 [B,S,H*16]) ----
#pragma unroll
    for (int r = 0; r < 8; ++r) {
        const float inv = (l8[r] > 0.f) ? 1.0f / l8[r] : 0.f;
#pragma unroll
        for (int t = 0; t < 5; ++t) o[t][r] *= inv;
    }
    const int sq = blockIdx.y * 128 + wave * 16;
#pragma unroll
    for (int t = 0; t < 5; ++t) {
        const int n = t * 16 + col16;
#pragma unroll
        for (int r = 0; r < 8; ++r) {
            const int s = sq + r + ((lane >> 4) << 3);
            const float v = o[t][r];
            if (n < 64)
                ctx[((size_t)b * S_ + s) * (H_ * 64) + h * 64 + n] = v;
            else
                lctx[((size_t)b * S_ + s) * (H_ * 16) + h * 16 + (n - 64)] = v;
        }
    }
}

// ---------------------------------------------------------------------------
// launch
// ---------------------------------------------------------------------------
extern "C" void kernel_launch(void* const* d_in, const int* in_sizes, int n_in,
                              void* d_out, int out_size, void* d_ws, size_t ws_size,
                              hipStream_t stream) {
    const float* hs  = (const float*)d_in[0];
    const float* li  = (const float*)d_in[1];
    const float* qw  = (const float*)d_in[2];
    const float* qb  = (const float*)d_in[3];
    const float* kw  = (const float*)d_in[4];
    const float* kb  = (const float*)d_in[5];
    const float* vw  = (const float*)d_in[6];
    const float* vb  = (const float*)d_in[7];
    const float* lqw = (const float*)d_in[8];
    const float* lqb = (const float*)d_in[9];
    const float* lkw = (const float*)d_in[10];
    const float* lkb = (const float*)d_in[11];
    const float* lvw = (const float*)d_in[12];
    const float* lvb = (const float*)d_in[13];

    const size_t M  = (size_t)B_ * S_;           // 8192
    const size_t BH = (size_t)B_ * H_ * S_;      // 98304 head-rows

    size_t off = 0;
    auto alloc = [&](size_t bytes) {
        size_t o = off; off += (bytes + 255) & ~(size_t)255; return o;
    };
    unsigned char* ws = (unsigned char*)d_ws;
    const size_t oHB  = alloc(M * D_ * 2);
    const size_t oLB  = alloc(M * DL_ * 2);
    const size_t oWQ  = alloc((size_t)D_ * D_ * 2);
    const size_t oWK  = alloc((size_t)D_ * D_ * 2);
    const size_t oWV  = alloc((size_t)D_ * D_ * 2);
    const size_t oWLQ = alloc((size_t)DL_ * DL_ * 2);
    const size_t oWLK = alloc((size_t)DL_ * DL_ * 2);
    const size_t oWLV = alloc((size_t)DL_ * DL_ * 2);
    const size_t oQA  = alloc(BH * AUG * 2);
    const size_t oKA  = alloc(BH * AUG * 2);
    const size_t oVA  = alloc(BH * VD * 2);
    (void)ws_size;

    unsigned short* HB  = (unsigned short*)(ws + oHB);
    unsigned short* LB  = (unsigned short*)(ws + oLB);
    unsigned short* WQ  = (unsigned short*)(ws + oWQ);
    unsigned short* WK  = (unsigned short*)(ws + oWK);
    unsigned short* WV  = (unsigned short*)(ws + oWV);
    unsigned short* WLQ = (unsigned short*)(ws + oWLQ);
    unsigned short* WLK = (unsigned short*)(ws + oWLK);
    unsigned short* WLV = (unsigned short*)(ws + oWLV);
    unsigned short* QAb = (unsigned short*)(ws + oQA);
    unsigned short* KAb = (unsigned short*)(ws + oKA);
    unsigned short* VAb = (unsigned short*)(ws + oVA);

    const dim3 blk(256);
    auto cvt = [&](const float* src, unsigned short* dst, int n) {
        cvt_bf16_kernel<<<dim3((n + 255) / 256), blk, 0, stream>>>(src, dst, n);
    };
    cvt(hs,  HB,  (int)(M * D_));
    cvt(li,  LB,  (int)(M * DL_));
    cvt(qw,  WQ,  D_ * D_);
    cvt(kw,  WK,  D_ * D_);
    cvt(vw,  WV,  D_ * D_);
    cvt(lqw, WLQ, DL_ * DL_);
    cvt(lkw, WLK, DL_ * DL_);
    cvt(lvw, WLV, DL_ * DL_);

    // zero QA/KA (covers the 16-element zero pad at dims 80..95)
    {
        const int nWords = (int)(2 * BH * AUG * 2 / 4);  // QA,KA contiguous
        zero_u32_kernel<<<dim3((nWords + 255) / 256), blk, 0, stream>>>(
            (unsigned int*)QAb, nWords);
    }

    // text projections -> augmented dims [0..63]
    const dim3 gT((unsigned)(M / 128), D_ / 64);
    gemm_proj_kernel<<<gT, blk, 0, stream>>>(HB, D_, WQ, qb, QAb,
                                             64, 1, AUG, S_ * AUG, 0, 1.f);
    gemm_proj_kernel<<<gT, blk, 0, stream>>>(HB, D_, WK, kb, KAb,
                                             64, 1, AUG, S_ * AUG, 0, 1.f);
    gemm_proj_kernel<<<gT, blk, 0, stream>>>(HB, D_, WV, vb, VAb,
                                             64, S_, 1, VD * S_, 0, 1.f);

    // layout projections -> augmented dims [64..79]; lq scaled by 2 so that
    // (q.k + (2lq).lk)/8 == q.k/8 + lq.lk/4
    const dim3 gL((unsigned)(M / 128), DL_ / 64);
    gemm_proj_kernel<<<gL, blk, 0, stream>>>(LB, DL_, WLQ, lqb, QAb,
                                             16, 1, AUG, S_ * AUG, 64, 2.f);
    gemm_proj_kernel<<<gL, blk, 0, stream>>>(LB, DL_, WLK, lkb, KAb,
                                             16, 1, AUG, S_ * AUG, 64, 1.f);
    gemm_proj_kernel<<<gL, blk, 0, stream>>>(LB, DL_, WLV, lvb, VAb,
                                             16, S_, 1, VD * S_, 64, 1.f);

    // fused attention
    float* ctx  = (float*)d_out;
    float* lctx = ctx + M * (size_t)D_;
    attn_kernel<<<dim3(B_ * H_, S_ / 128), blk, 0, stream>>>(QAb, KAb, VAb,
                                                             ctx, lctx);
    (void)in_sizes; (void)n_in; (void)out_size;
}
